// VectorQuantizer_73048803770683
// MI455X (gfx1250) — matrix operand order repaired and checked by reading, not compiled, not run on hardware
//
#include <hip/hip_runtime.h>

typedef __attribute__((ext_vector_type(16))) _Float16 v16h;
typedef __attribute__((ext_vector_type(8)))  _Float16 v8h;
typedef __attribute__((ext_vector_type(8)))  float    v8f;

#define DIM      64
#define KCODES   1024
#define NROWS    (16 * 8192)            // 131072
#define LSTRIDE  72                     // halves per code row in LDS (64 + 8 pad -> 144B, 16B aligned, bank-tiling)
#define QELEMS   (NROWS * DIM)          // 8388608

// ---------------------------------------------------------------------------
// Kernel 1: transpose embeddings [D,K] (f32) -> embT [K,D] (f32) in scratch,
// so the per-row quantized gather is a coalesced contiguous 256B read.
// ---------------------------------------------------------------------------
__global__ __launch_bounds__(256)
void vq_transpose_kernel(const float* __restrict__ emb, float* __restrict__ embT) {
    int o = blockIdx.x * 256 + threadIdx.x;   // 65536 threads total
    int k = o >> 6;
    int d = o & 63;
    embT[o] = emb[d * KCODES + k];
}

// ---------------------------------------------------------------------------
// Kernel 2: main VQ kernel. 1024 workgroups x 256 threads (8 waves).
// Each wave handles 16 rows; WG handles 128 rows.
// ---------------------------------------------------------------------------
__global__ __launch_bounds__(256)
void vq_main_kernel(const float* __restrict__ x,
                    const float* __restrict__ emb,
                    const float* __restrict__ embT,
                    float* __restrict__ qout,
                    int*   __restrict__ iout,
                    float* __restrict__ wgpart) {
    __shared__ _Float16 sT[KCODES * LSTRIDE];   // f16 codebook, [k][d], padded rows (144 KB)
    __shared__ float    sE2[KCODES];            // ||e_k||^2 in f32 (4 KB)
    __shared__ float    sPart[8];

    const int t = threadIdx.x;

    // --- Fill f16 transposed codebook into LDS (coalesced global reads). ---
    #pragma unroll 4
    for (int i = 0; i < 256; ++i) {
        int e = t + 256 * i;          // e = d*1024 + k, consecutive threads -> consecutive k
        int d = e >> 10;
        int k = e & 1023;
        sT[k * LSTRIDE + d] = (_Float16)emb[e];
    }
    // --- e2[k] = sum_d emb[d][k]^2 in f32 (coalesced across lanes in k). ---
    #pragma unroll
    for (int j = 0; j < 4; ++j) {
        int k = t + 256 * j;
        float s = 0.0f;
        #pragma unroll 8
        for (int d = 0; d < DIM; ++d) {
            float v = emb[d * KCODES + k];
            s = fmaf(v, v, s);
        }
        sE2[k] = s;
    }
    __syncthreads();

    const int lane    = t & 31;
    const int wave    = t >> 5;
    const int rowBase = blockIdx.x * 128 + wave * 16;
    const int mrow    = rowBase + (lane & 15);
    const int hi      = (lane >= 16) ? 1 : 0;

    // --- Build A tiles (16x64 f16) per the CDNA5 16-bit A layout. ---
    // a0 covers d=0..31, a1 covers d=32..63.
    // lane<16: halves 0..7 -> d=0..7,  halves 8..15 -> d=16..23 (a0)
    // lane>=16: halves 0..7 -> d=8..15, halves 8..15 -> d=24..31 (a0); a1 = +32
    const float* xr = x + (long)mrow * DIM;
    const int base0 = hi * 8;
    v16h a0, a1;
    #pragma unroll
    for (int q = 0; q < 2; ++q) {                 // q: half-group (halves 8q..8q+7)
        int dA = base0 + q * 16;                  // for a0
        float4 f0 = *(const float4*)(xr + dA);
        float4 f1 = *(const float4*)(xr + dA + 4);
        a0[8*q + 0] = (_Float16)f0.x;  a0[8*q + 1] = (_Float16)f0.y;
        a0[8*q + 2] = (_Float16)f0.z;  a0[8*q + 3] = (_Float16)f0.w;
        a0[8*q + 4] = (_Float16)f1.x;  a0[8*q + 5] = (_Float16)f1.y;
        a0[8*q + 6] = (_Float16)f1.z;  a0[8*q + 7] = (_Float16)f1.w;
        float4 g0 = *(const float4*)(xr + 32 + dA);
        float4 g1 = *(const float4*)(xr + 32 + dA + 4);
        a1[8*q + 0] = (_Float16)g0.x;  a1[8*q + 1] = (_Float16)g0.y;
        a1[8*q + 2] = (_Float16)g0.z;  a1[8*q + 3] = (_Float16)g0.w;
        a1[8*q + 4] = (_Float16)g1.x;  a1[8*q + 5] = (_Float16)g1.y;
        a1[8*q + 6] = (_Float16)g1.z;  a1[8*q + 7] = (_Float16)g1.w;
    }

    // --- Chunk loop over 1024 codes, 32 at a time (two interleaved 16-code
    //     chunks so chunk-1's WMMAs cover chunk-0's WMMA->VALU hazard). ---
    float bv[8];
    int   bi[8];
    #pragma unroll
    for (int r = 0; r < 8; ++r) { bv[r] = 3.4e38f; bi[r] = 0; }

    const int col = lane & 15;
    const int dhi = hi * 8;                       // B layout: lane<16 -> d 0..7,16..23, lane>=16 -> d 8..15,24..31

    for (int c = 0; c < 64; c += 2) {
        const int k0 = c * 16 + col;
        const int k1 = k0 + 16;
        const _Float16* bp0 = &sT[k0 * LSTRIDE + dhi];
        const _Float16* bp1 = &sT[k1 * LSTRIDE + dhi];

        union { v16h v; v8h h[2]; } b00, b01, b10, b11;
        b00.h[0] = *(const v8h*)(bp0);            // d = dhi .. dhi+7
        b00.h[1] = *(const v8h*)(bp0 + 16);       // d = dhi+16 .. dhi+23
        b01.h[0] = *(const v8h*)(bp0 + 32);       // d = 32+dhi ..
        b01.h[1] = *(const v8h*)(bp0 + 48);
        b10.h[0] = *(const v8h*)(bp1);
        b10.h[1] = *(const v8h*)(bp1 + 16);
        b11.h[0] = *(const v8h*)(bp1 + 32);
        b11.h[1] = *(const v8h*)(bp1 + 48);

        v8f acc0 = {};
        v8f acc1 = {};
        acc0 = __builtin_amdgcn_wmma_f32_16x16x32_f16(false, a0, false, b00.v,
                                                      (short)0, acc0, false, false);
        acc0 = __builtin_amdgcn_wmma_f32_16x16x32_f16(false, a1, false, b01.v,
                                                      (short)0, acc0, false, false);
        acc1 = __builtin_amdgcn_wmma_f32_16x16x32_f16(false, a0, false, b10.v,
                                                      (short)0, acc1, false, false);
        acc1 = __builtin_amdgcn_wmma_f32_16x16x32_f16(false, a1, false, b11.v,
                                                      (short)0, acc1, false, false);

        const float e20 = sE2[k0];
        const float e21 = sE2[k1];
        #pragma unroll
        for (int r = 0; r < 8; ++r) {
            float s = fmaf(-2.0f, acc0[r], e20);  // ||x||^2 omitted: constant per row
            if (s < bv[r]) { bv[r] = s; bi[r] = k0; }
        }
        #pragma unroll
        for (int r = 0; r < 8; ++r) {
            float s = fmaf(-2.0f, acc1[r], e21);
            if (s < bv[r]) { bv[r] = s; bi[r] = k1; }
        }
    }

    // --- Reduce argmin across the 16 lanes of each half (argmin tie: lowest idx). ---
    #pragma unroll
    for (int r = 0; r < 8; ++r) {
        #pragma unroll
        for (int msk = 1; msk < 16; msk <<= 1) {
            float ov = __shfl_xor(bv[r], msk, 32);
            int   oi = __shfl_xor(bi[r], msk, 32);
            if (ov < bv[r] || (ov == bv[r] && oi < bi[r])) { bv[r] = ov; bi[r] = oi; }
        }
    }

    // Rows 0..7 live in lanes 0..15, rows 8..15 in lanes 16..31 (all lanes of a half agree).
    if (lane == 0) {
        #pragma unroll
        for (int r = 0; r < 8; ++r) iout[rowBase + r] = bi[r];
    }
    if (lane == 16) {
        #pragma unroll
        for (int r = 0; r < 8; ++r) iout[rowBase + 8 + r] = bi[r];
    }

    // --- Gather quantized rows (f32, coalesced from embT) + loss accumulation. ---
    float lsum = 0.0f;
    #pragma unroll
    for (int mm = 0; mm < 16; ++mm) {
        const int src = (mm < 8) ? 0 : 16;
        int km = __shfl(bi[mm & 7], src, 32);
        long row = rowBase + mm;
        float q0 = embT[km * DIM + lane];
        float q1 = embT[km * DIM + lane + 32];
        float x0 = x[row * DIM + lane];
        float x1 = x[row * DIM + lane + 32];
        qout[row * DIM + lane]      = q0;
        qout[row * DIM + lane + 32] = q1;
        float d0 = q0 - x0, d1 = q1 - x1;
        lsum = fmaf(d0, d0, lsum);
        lsum = fmaf(d1, d1, lsum);
    }
    #pragma unroll
    for (int msk = 1; msk < 32; msk <<= 1) lsum += __shfl_xor(lsum, msk, 32);
    if (lane == 0) sPart[wave] = lsum;
    __syncthreads();
    if (t == 0) {
        float s = 0.0f;
        #pragma unroll
        for (int w = 0; w < 8; ++w) s += sPart[w];
        wgpart[blockIdx.x] = s;
    }
}

// ---------------------------------------------------------------------------
// Kernel 3: deterministic fixed-order reduction of 1024 WG partials -> loss.
// loss = (BETA + 1) * mean((q - x)^2) = 1.25 * sum / (N*D)
// ---------------------------------------------------------------------------
__global__ __launch_bounds__(256)
void vq_loss_kernel(const float* __restrict__ wgpart, float* __restrict__ lossOut) {
    __shared__ float sm[256];
    int t = threadIdx.x;
    float s = wgpart[t] + wgpart[t + 256] + wgpart[t + 512] + wgpart[t + 768];
    sm[t] = s;
    __syncthreads();
    for (int st = 128; st > 0; st >>= 1) {
        if (t < st) sm[t] += sm[t + st];
        __syncthreads();
    }
    if (t == 0) lossOut[0] = sm[0] * (1.25f / (float)QELEMS);
}

// ---------------------------------------------------------------------------
extern "C" void kernel_launch(void* const* d_in, const int* in_sizes, int n_in,
                              void* d_out, int out_size, void* d_ws, size_t ws_size,
                              hipStream_t stream) {
    const float* x   = (const float*)d_in[0];   // [16, 8192, 64] f32
    const float* emb = (const float*)d_in[1];   // [64, 1024] f32

    float* qout = (float*)d_out;                        // 8388608 f32
    int*   iout = (int*)((float*)d_out + QELEMS);       // 131072 i32 (bitwise)
    float* lout = (float*)d_out + QELEMS + NROWS;       // 1 f32

    float* embT   = (float*)d_ws;                       // 65536 f32
    float* wgpart = embT + (DIM * KCODES);              // 1024 f32

    vq_transpose_kernel<<<256, 256, 0, stream>>>(emb, embT);
    vq_main_kernel<<<1024, 256, 0, stream>>>(x, emb, embT, qout, iout, wgpart);
    vq_loss_kernel<<<1, 256, 0, stream>>>(wgpart, lout);
}
